// LSTM_9474697855554
// MI455X (gfx1250) — matrix operand order repaired and checked by reading, not compiled, not run on hardware
//
#include <hip/hip_runtime.h>
#include <math.h>

// LSTM: B=128, T=512, D=1024, H=1024.
// z = [x_t | h_{t-1}] @ [Wf|Wi|Wo|Wc] + b -> gates -> c,h.
// bf16 WMMA (v_wmma_f32_16x16x32_bf16), f32 accumulate.
// - Weights pre-swizzled ONCE into WMMA-fragment-major layout (16MB, L2-resident):
//   B fragments are direct global_load_b128 pairs (address-space-1 pointers,
//   per-gate 64-bit pointer induction).
// - A operand kept as a contiguous per-step [128][2048] bf16 ping/pong buffer:
//   cols 0..1023 = x_t (converted by the PREVIOUS step's epilogue),
//   cols 1024..2047 = h_{t-1} (written by the previous step's epilogue).
// - A tile staged in LDS fragment-major (ds_load_b128 pairs), double-buffered.

#define B_SZ 128
#define T_SZ 512
#define D_SZ 1024
#define H_SZ 1024
#define K_SZ (D_SZ + H_SZ) // 2048
#define NKT  (K_SZ / 32)   // 64 k-tiles

typedef __bf16 bf16_t;
typedef bf16_t         v16bf __attribute__((ext_vector_type(16)));
typedef float          v8f   __attribute__((ext_vector_type(8)));
typedef unsigned short u16x8 __attribute__((ext_vector_type(8)));
typedef float          f32x4 __attribute__((ext_vector_type(4)));

// Global (address space 1) pointer types: force global_load_* (not flat_load_*,
// which would tie up DScnt and couple waits with the LDS ping-pong).
typedef const u16x8 __attribute__((address_space(1)))* gc_u16x8p;

union frag_u { u16x8 h[2]; v16bf v; };

__device__ __forceinline__ gc_u16x8p to_global(const u16x8* p) {
    return (gc_u16x8p)(unsigned long long)p;
}

__device__ __forceinline__ unsigned short f32_to_bf16_bits(float f) {
    union { float f; unsigned int u; } v; v.f = f;
    unsigned int u = v.u;
    return (unsigned short)((u + 0x7FFFu + ((u >> 16) & 1u)) >> 16); // RNE
}

// fast sigmoid / tanh: v_exp_f32 + v_rcp_f32
__device__ __forceinline__ float fast_sigmoid(float y) {
    return __builtin_amdgcn_rcpf(1.0f + __expf(-y));
}
__device__ __forceinline__ float fast_tanh(float x) {
    return 2.0f * fast_sigmoid(2.0f * x) - 1.0f;
}

// ---- One-time: fp32 weights -> bf16, swizzled fragment-major ----
// tile (g,kt,ct) = 32k x 64n; offset = tileIdx*2048 + nb*512 + lane*16 + e
// (ushort units), lane = (n%16)+16*(k%32/16), e = k%16.
__global__ __launch_bounds__(256)
void lstm_convert_weights(const float* __restrict__ Wf,
                          const float* __restrict__ Wi,
                          const float* __restrict__ Wo,
                          const float* __restrict__ Wc,
                          unsigned short* __restrict__ dst) {
    const int per = K_SZ * H_SZ;
    int idx = blockIdx.x * blockDim.x + threadIdx.x;
    if (idx >= 4 * per) return;
    int g   = idx / per;
    int rem = idx - g * per;
    const float* src = (g == 0) ? Wf : (g == 1) ? Wi : (g == 2) ? Wo : Wc;
    unsigned short val = f32_to_bf16_bits(src[rem]);

    int k  = rem >> 10;
    int n  = rem & 1023;
    int kt = k >> 5, kk = k & 31;
    int ct = n >> 6, nc = n & 63;
    int nb = nc >> 4, ln = nc & 15;
    int lh = kk >> 4, e = kk & 15;
    size_t off = (size_t)((g * NKT + kt) * 16 + ct) * 2048
               + (size_t)nb * 512 + (size_t)(ln + 16 * lh) * 16 + e;
    dst[off] = val;
}

// ---- One-time: seed A0 = [x_0 | 0] (bf16). 16 elems per thread. ----
__global__ __launch_bounds__(256)
void lstm_init_A(const float* __restrict__ x, unsigned short* __restrict__ A0) {
    size_t base = ((size_t)blockIdx.x * blockDim.x + threadIdx.x) * 16;
    if (base >= (size_t)B_SZ * K_SZ) return;
    int row = (int)(base >> 11);
    int col = (int)(base & (K_SZ - 1));
    u16x8 r0 = (u16x8){}, r1 = (u16x8){};
    if (col < D_SZ) { // x region: t=0 slice
        const float* s = x + (size_t)row * (T_SZ * D_SZ) + col;
#pragma unroll
        for (int i = 0; i < 2; ++i) {
            f32x4 f = ((const f32x4*)s)[i];
            r0[i*4+0] = f32_to_bf16_bits(f[0]); r0[i*4+1] = f32_to_bf16_bits(f[1]);
            r0[i*4+2] = f32_to_bf16_bits(f[2]); r0[i*4+3] = f32_to_bf16_bits(f[3]);
        }
#pragma unroll
        for (int i = 2; i < 4; ++i) {
            f32x4 f = ((const f32x4*)s)[i];
            r1[(i-2)*4+0] = f32_to_bf16_bits(f[0]); r1[(i-2)*4+1] = f32_to_bf16_bits(f[1]);
            r1[(i-2)*4+2] = f32_to_bf16_bits(f[2]); r1[(i-2)*4+3] = f32_to_bf16_bits(f[3]);
        }
    }
    *(u16x8*)(A0 + base)     = r0;
    *(u16x8*)(A0 + base + 8) = r1;
}

// ---- One-time: zero cell state ----
__global__ __launch_bounds__(256)
void lstm_init_c(float* __restrict__ c) {
    int idx = blockIdx.x * blockDim.x + threadIdx.x;
    if (idx < B_SZ * H_SZ) c[idx] = 0.0f;
}

// ---- One LSTM timestep ----
// Grid (H/64, B/64), 256 threads (8 waves). Wave w: row block (w>>1)*16,
// col half (w&1)*32; computes 16x32 for ALL 4 gates -> epilogue in registers.
__global__ __launch_bounds__(256)
void lstm_step(int t,
               const float* __restrict__ x,            // [B,T,D] f32 (t+1 slice)
               const unsigned short* __restrict__ Wb,  // swizzled bf16 weights
               const float* __restrict__ bf_, const float* __restrict__ bi_,
               const float* __restrict__ bo_, const float* __restrict__ bc_,
               const unsigned short* __restrict__ Ain, // [B][2048] bf16: x_t | h_{t-1}
               unsigned short* __restrict__ Anext,     // [B][2048] bf16: x_{t+1} | h_t
               float* __restrict__ cst,                // [B,H] f32
               float* __restrict__ out)                // [B,T,H] f32 (reversed t)
{
    __shared__ u16x8 Asw[2][4][32][2]; // [buf][rowblock][lane][chunk]

    const int tid  = threadIdx.x;
    const int wave = tid >> 5;
    const int lane = tid & 31;
    const int ln   = lane & 15;
    const int lh   = lane >> 4;

    const int ct  = blockIdx.x;
    const int C0  = ct * 64;
    const int R0  = blockIdx.y * 64;
    const int rbW = wave >> 1;
    const int m0  = rbW * 16;
    const int nb0 = (wave & 1) * 2;

    // A staging: thread -> one u16x8 chunk (8 consecutive k) of the 64x32 tile.
    const int arow    = tid >> 2;
    const int aq      = tid & 3;
    const int a_rb    = arow >> 4;
    const int a_lane  = (arow & 15) + 16 * (aq >> 1);
    const int a_chunk = aq & 1;

    // A-source pointer: single contiguous stride, +4 u16x8 (32 elems) per k-tile.
    gc_u16x8p pA = to_global(
        (const u16x8*)(Ain + (size_t)(R0 + arow) * K_SZ + (aq * 8)));

    v8f acc[4][2];
#pragma unroll
    for (int g = 0; g < 4; ++g)
#pragma unroll
        for (int cb = 0; cb < 2; ++cb)
            acc[g][cb] = (v8f){};

    u16x8 aReg = pA[0];
    Asw[0][a_rb][a_lane][a_chunk] = aReg;
    __syncthreads();

    // Per-gate B pointers (global AS): +4096 u16x8 (64KB) per k-tile; nb/chunk
    // fold into the 24-bit instruction offsets.
    gc_u16x8p pB[4];
#pragma unroll
    for (int g = 0; g < 4; ++g)
        pB[g] = to_global((const u16x8*)Wb + (size_t)g * (NKT * 16 * 256)
                          + (size_t)ct * 256 + (size_t)lane * 2);

    for (int kt = 0; kt < NKT; ++kt) {
        const int cur = kt & 1;
        const bool havenext = (kt + 1 < NKT);

        if (havenext) aReg = pA[(kt + 1) * 4];

        frag_u fa;                       // 2x ds_load_b128
        fa.h[0] = Asw[cur][rbW][lane][0];
        fa.h[1] = Asw[cur][rbW][lane][1];

#pragma unroll
        for (int g = 0; g < 4; ++g) {
#pragma unroll
            for (int cb = 0; cb < 2; ++cb) {
                const int nb = nb0 + cb;
                frag_u fb;                // 2x global_load_b128 (L2-resident)
                fb.h[0] = pB[g][nb * 64 + 0];
                fb.h[1] = pB[g][nb * 64 + 1];
                acc[g][cb] = __builtin_amdgcn_wmma_f32_16x16x32_bf16(
                    false, fa.v, false, fb.v, (short)0, acc[g][cb], false, false);
            }
            pB[g] += 4096;
        }

        if (havenext) {
            Asw[cur ^ 1][a_rb][a_lane][a_chunk] = aReg;
            __syncthreads();
        }
    }

    // ---- Epilogue 1: gates + state update, h_t -> Anext[:,1024+col] ----
#pragma unroll
    for (int cb = 0; cb < 2; ++cb) {
#pragma unroll
        for (int r = 0; r < 8; ++r) {
            int col = C0 + (nb0 + cb) * 16 + ln;
            int row = R0 + m0 + r + 8 * lh;
            float zf = acc[0][cb][r] + bf_[col];
            float zi = acc[1][cb][r] + bi_[col];
            float zo = acc[2][cb][r] + bo_[col];
            float zg = acc[3][cb][r] + bc_[col];
            float fg = fast_sigmoid(zf);
            float ig = fast_sigmoid(zi);
            float og = fast_sigmoid(zo);
            float gg = fast_tanh(zg);
            size_t ci = (size_t)row * H_SZ + col;
            float cn = fg * cst[ci] + ig * gg;
            cst[ci]  = cn;
            float h  = og * fast_tanh(cn);
            Anext[(size_t)row * K_SZ + D_SZ + col] = f32_to_bf16_bits(h);
            // out is written once, never re-read: nontemporal, spare the L2
            __builtin_nontemporal_store(
                h, out + (size_t)row * (T_SZ * H_SZ)
                       + (size_t)(T_SZ - 1 - t) * H_SZ + col);
        }
    }

    // ---- Epilogue 2: prefetch-convert x_{t+1} 64x64 patch into Anext ----
    if (t + 1 < T_SZ) {
        const int xr  = tid >> 2;            // 0..63
        const int xc0 = (tid & 3) * 16;      // 0,16,32,48
        const float* s = x + (size_t)(R0 + xr) * (T_SZ * D_SZ)
                       + (size_t)(t + 1) * D_SZ + C0 + xc0;
        u16x8 r0, r1;
#pragma unroll
        for (int i = 0; i < 2; ++i) {
            f32x4 f = __builtin_nontemporal_load((const f32x4*)s + i);
            r0[i*4+0] = f32_to_bf16_bits(f[0]); r0[i*4+1] = f32_to_bf16_bits(f[1]);
            r0[i*4+2] = f32_to_bf16_bits(f[2]); r0[i*4+3] = f32_to_bf16_bits(f[3]);
        }
#pragma unroll
        for (int i = 2; i < 4; ++i) {
            f32x4 f = __builtin_nontemporal_load((const f32x4*)s + i);
            r1[(i-2)*4+0] = f32_to_bf16_bits(f[0]); r1[(i-2)*4+1] = f32_to_bf16_bits(f[1]);
            r1[(i-2)*4+2] = f32_to_bf16_bits(f[2]); r1[(i-2)*4+3] = f32_to_bf16_bits(f[3]);
        }
        unsigned short* d = Anext + (size_t)(R0 + xr) * K_SZ + C0 + xc0;
        *(u16x8*)d       = r0;
        *(u16x8*)(d + 8) = r1;
    }
}

extern "C" void kernel_launch(void* const* d_in, const int* in_sizes, int n_in,
                              void* d_out, int out_size, void* d_ws, size_t ws_size,
                              hipStream_t stream) {
    (void)in_sizes; (void)n_in; (void)out_size; (void)ws_size;
    const float* x  = (const float*)d_in[0];
    const float* Wf = (const float*)d_in[1];
    const float* bf = (const float*)d_in[2];
    const float* Wi = (const float*)d_in[3];
    const float* bi = (const float*)d_in[4];
    const float* Wo = (const float*)d_in[5];
    const float* bo = (const float*)d_in[6];
    const float* Wc = (const float*)d_in[7];
    const float* bc = (const float*)d_in[8];
    float* out = (float*)d_out;

    // Workspace: 16MB swizzled weights | A ping (512KB) | A pong (512KB) | c (512KB)
    const size_t wbytes = (size_t)4 * K_SZ * H_SZ * 2;
    const size_t abytes = (size_t)B_SZ * K_SZ * 2;
    char* ws = (char*)d_ws;
    unsigned short* Wb = (unsigned short*)ws;
    unsigned short* A0 = (unsigned short*)(ws + wbytes);
    unsigned short* A1 = (unsigned short*)(ws + wbytes + abytes);
    float*         cst = (float*)(ws + wbytes + 2 * abytes);

    lstm_convert_weights<<<(4 * K_SZ * H_SZ) / 256, 256, 0, stream>>>(Wf, Wi, Wo, Wc, Wb);
    lstm_init_A<<<((B_SZ * K_SZ) / 16 + 255) / 256, 256, 0, stream>>>(x, A0);
    lstm_init_c<<<(B_SZ * H_SZ) / 256, 256, 0, stream>>>(cst);

    dim3 grid(H_SZ / 64, B_SZ / 64); // 16 x 2 workgroups
    for (int t = 0; t < T_SZ; ++t) {
        unsigned short* Ain   = (t & 1) ? A1 : A0;
        unsigned short* Anext = (t & 1) ? A0 : A1;
        lstm_step<<<grid, 256, 0, stream>>>(t, x, Wb, bf, bi, bo, bc,
                                            Ain, Anext, cst, out);
    }
}
